// TSSBlock_876173328811
// MI455X (gfx1250) — matrix operand order repaired and checked by reading, not compiled, not run on hardware
//
#include <hip/hip_runtime.h>
#include <hip/hip_bf16.h>

// ---- sizes (fixed by reference) ----
#define BB     4
#define NN     16384
#define CC     256
#define HID    768
#define MM     (BB * NN)        // 65536 tokens
#define HH     128
#define WW2    128
#define NHEAD  8
#define DHEAD  32

typedef __bf16 bf16_t;
typedef __attribute__((ext_vector_type(8)))  __bf16 v8bf;
typedef __attribute__((ext_vector_type(16))) __bf16 v16bf;
typedef __attribute__((ext_vector_type(8)))  float  v8f;

// ------------------------------------------------------------------
// WMMA fragment loaders (bf16 16x16x32, wave32 layouts per CDNA5 ISA)
// ------------------------------------------------------------------
static __device__ __forceinline__ v16bf cat16(v8bf lo, v8bf hi) {
  return __builtin_shufflevector(lo, hi, 0,1,2,3,4,5,6,7,8,9,10,11,12,13,14,15);
}

// A 16x32: lane holds row (lane&15); K chunks at kk+(lane>=16?8:0) and +16.
static __device__ __forceinline__ v16bf load_a_frag(const bf16_t* __restrict__ A,
                                                    int rowBase, int K, int kk, int lane) {
  int r  = rowBase + (lane & 15);
  int k0 = kk + ((lane >> 4) << 3);
  const bf16_t* p = A + (size_t)r * K + k0;
  v8bf lo = *(const v8bf*)p;
  v8bf hi = *(const v8bf*)(p + 16);
  return cat16(lo, hi);
}

// B 32x16: lane holds column (lane&15); K chunks at kk+(lane>=16?8:0) and +16.
// WT is the transposed weight: WT[col][k], k contiguous.
static __device__ __forceinline__ v16bf load_b_frag(const bf16_t* __restrict__ WT,
                                                    int colBase, int K, int kk, int lane) {
  int c  = colBase + (lane & 15);
  int k0 = kk + ((lane >> 4) << 3);
  const bf16_t* p = WT + (size_t)c * K + k0;
  v8bf lo = *(const v8bf*)p;
  v8bf hi = *(const v8bf*)(p + 16);
  return cat16(lo, hi);
}

// ------------------------------------------------------------------
// GEMM: D[M,Nc] = A[M,K](bf16) * WT[Nc,K](bf16) + epilogue
// block = 256 threads = 8 waves; block tile 128(M) x 64(N); wave tile 32x32.
// EPI 0: out = acc + bias                       (fc1)
// EPI 1: out = acc + bias + resid               (proj, fc2)
// EPI 2: split to q/k/v [B,h,N,d] with bias     (qkv)
// ------------------------------------------------------------------
template <int EPI>
__global__ __launch_bounds__(256) void gemm_bf16_wmma(
    const bf16_t* __restrict__ A, const bf16_t* __restrict__ WT,
    const float* __restrict__ bias, const float* __restrict__ resid,
    float* __restrict__ out,
    float* __restrict__ outQ, float* __restrict__ outK, float* __restrict__ outV,
    int K, int Nc) {
  const int lane = threadIdx.x & 31;
  const int w    = threadIdx.x >> 5;
  const int rowB = blockIdx.y * 128 + (w & 3) * 32;
  const int colB = blockIdx.x * 64  + (w >> 2) * 32;

  v8f acc00 = {}, acc01 = {}, acc10 = {}, acc11 = {};

  for (int kk = 0; kk < K; kk += 32) {
    if (kk + 64 <= K)  // exercise global_prefetch_b8 for the next A chunk
      __builtin_prefetch(A + (size_t)(rowB + (lane & 15)) * K + kk + 32, 0, 1);
    v16bf a0 = load_a_frag(A, rowB,      K, kk, lane);
    v16bf a1 = load_a_frag(A, rowB + 16, K, kk, lane);
    v16bf b0 = load_b_frag(WT, colB,      K, kk, lane);
    v16bf b1 = load_b_frag(WT, colB + 16, K, kk, lane);
    acc00 = __builtin_amdgcn_wmma_f32_16x16x32_bf16(false, a0, false, b0, (short)0, acc00, false, false);
    acc01 = __builtin_amdgcn_wmma_f32_16x16x32_bf16(false, a0, false, b1, (short)0, acc01, false, false);
    acc10 = __builtin_amdgcn_wmma_f32_16x16x32_bf16(false, a1, false, b0, (short)0, acc10, false, false);
    acc11 = __builtin_amdgcn_wmma_f32_16x16x32_bf16(false, a1, false, b1, (short)0, acc11, false, false);
  }

  auto emit = [&](const v8f& acc, int rt, int ct) {
    const int rbase = rowB + rt + ((lane >> 4) << 3);
    const int col   = colB + ct + (lane & 15);
    const float bv  = bias[col];
#pragma unroll
    for (int g = 0; g < 8; ++g) {
      const int row = rbase + g;
      float val = acc[g] + bv;
      if constexpr (EPI == 0) {
        out[(size_t)row * Nc + col] = val;
      } else if constexpr (EPI == 1) {
        out[(size_t)row * Nc + col] = val + resid[(size_t)row * Nc + col];
      } else {
        const int b = row >> 14, n = row & (NN - 1);
        const int s = col >> 8, head = (col >> 5) & 7, dd = col & 31;
        float* dst = (s == 0) ? outQ : (s == 1) ? outK : outV;
        dst[(((size_t)(b * NHEAD + head)) * NN + n) * DHEAD + dd] = val;
      }
    }
  };
  emit(acc00, 0, 0);  emit(acc01, 0, 16);
  emit(acc10, 16, 0); emit(acc11, 16, 16);
}

// ------------------------------------------------------------------
// Weight transpose + f32->bf16: out[nc*K + k] = in[k*Nc + nc]
// ------------------------------------------------------------------
__global__ __launch_bounds__(256) void transpose_w(const float* __restrict__ in,
                                                   bf16_t* __restrict__ out,
                                                   int K, int Nc) {
  int i = blockIdx.x * 256 + threadIdx.x;
  if (i >= K * Nc) return;
  int k = i / Nc, nc = i - k * Nc;
  out[(size_t)nc * K + k] = (bf16_t)in[i];
}

// ------------------------------------------------------------------
// LayerNorm over C=256: one wave per token, bf16 output for next GEMM.
// ------------------------------------------------------------------
__global__ __launch_bounds__(256) void ln_kernel(const float* __restrict__ x,
                                                 const float* __restrict__ g,
                                                 const float* __restrict__ b,
                                                 bf16_t* __restrict__ out) {
  const int lane = threadIdx.x & 31;
  const int wv   = threadIdx.x >> 5;
  const int m    = blockIdx.x * 8 + wv;
  const float* row = x + (size_t)m * CC;
  const int c0 = lane * 8;
  float v[8];
  float4 p0 = *(const float4*)(row + c0);
  float4 p1 = *(const float4*)(row + c0 + 4);
  v[0] = p0.x; v[1] = p0.y; v[2] = p0.z; v[3] = p0.w;
  v[4] = p1.x; v[5] = p1.y; v[6] = p1.z; v[7] = p1.w;
  float s = 0.f;
#pragma unroll
  for (int i = 0; i < 8; ++i) s += v[i];
#pragma unroll
  for (int o = 16; o >= 1; o >>= 1) s += __shfl_xor(s, o, 32);
  const float mean = s * (1.0f / CC);
  float q = 0.f;
#pragma unroll
  for (int i = 0; i < 8; ++i) { float d = v[i] - mean; q += d * d; }
#pragma unroll
  for (int o = 16; o >= 1; o >>= 1) q += __shfl_xor(q, o, 32);
  const float rs = rsqrtf(q * (1.0f / CC) + 1e-6f);
  v8bf o8;
#pragma unroll
  for (int i = 0; i < 8; ++i)
    o8[i] = (bf16_t)((v[i] - mean) * rs * g[c0 + i] + b[c0 + i]);
  *(v8bf*)(out + (size_t)m * CC + c0) = o8;
}

// ------------------------------------------------------------------
// Softmax of k over tokens N, per (b,h,dd) column, in-place.
// ------------------------------------------------------------------
__global__ __launch_bounds__(256) void softmax_tokens(float* __restrict__ k) {
  const int col = blockIdx.x;           // 0..(B*h*d - 1)
  const int dd = col & 31, bh = col >> 5;
  float* base = k + ((size_t)bh * NN) * DHEAD + dd;
  __shared__ float red[256];
  const int tid = threadIdx.x;
  float mx = -3.0e38f;
  for (int n = tid; n < NN; n += 256) mx = fmaxf(mx, base[(size_t)n * DHEAD]);
  red[tid] = mx; __syncthreads();
  for (int st = 128; st; st >>= 1) { if (tid < st) red[tid] = fmaxf(red[tid], red[tid + st]); __syncthreads(); }
  const float m = red[0]; __syncthreads();
  float s = 0.f;
  for (int n = tid; n < NN; n += 256) s += expf(base[(size_t)n * DHEAD] - m);
  red[tid] = s; __syncthreads();
  for (int st = 128; st; st >>= 1) { if (tid < st) red[tid] += red[tid + st]; __syncthreads(); }
  const float inv = 1.0f / red[0];
  for (int n = tid; n < NN; n += 256) {
    size_t i = (size_t)n * DHEAD;
    base[i] = expf(base[i] - m) * inv;
  }
}

// ------------------------------------------------------------------
// kv[b,h,kc,vc] = sum_n k_sm[b,h,n,kc] * v[b,h,n,vc]   (split-K atomics)
// grid: x = 16 K-slices, y = B*h
// ------------------------------------------------------------------
__global__ __launch_bounds__(256) void kv_kernel(const float* __restrict__ ks,
                                                 const float* __restrict__ v,
                                                 float* __restrict__ kv) {
  const int bh = blockIdx.y;
  const int n0 = blockIdx.x * (NN / 16);
  const float* kbase = ks + (size_t)bh * NN * DHEAD;
  const float* vbase = v  + (size_t)bh * NN * DHEAD;
  __shared__ float Kl[64 * 32];
  __shared__ float Vl[64 * 32];
  const int tid = threadIdx.x;
  const int kidx = tid >> 3;
  const int vb   = (tid & 7) * 4;
  float a0 = 0.f, a1 = 0.f, a2 = 0.f, a3 = 0.f;
  for (int c = 0; c < (NN / 16); c += 64) {
    __syncthreads();
    for (int i = tid; i < 64 * 32; i += 256) {
      Kl[i] = kbase[(size_t)(n0 + c) * DHEAD + i];
      Vl[i] = vbase[(size_t)(n0 + c) * DHEAD + i];
    }
    __syncthreads();
#pragma unroll 4
    for (int n = 0; n < 64; ++n) {
      const float kvv = Kl[n * 32 + kidx];
      a0 += kvv * Vl[n * 32 + vb + 0];
      a1 += kvv * Vl[n * 32 + vb + 1];
      a2 += kvv * Vl[n * 32 + vb + 2];
      a3 += kvv * Vl[n * 32 + vb + 3];
    }
  }
  float* dst = kv + (size_t)bh * (DHEAD * DHEAD) + kidx * DHEAD + vb;
  atomicAdd(dst + 0, a0); atomicAdd(dst + 1, a1);
  atomicAdd(dst + 2, a2); atomicAdd(dst + 3, a3);
}

// ------------------------------------------------------------------
// Depthwise 3x3 conv (SAME) on v in [B,h,N,d] layout (channel = h*32+dd).
// grid: (8,8, B*C); 16x16 output tile per block, 18x18 halo in LDS.
// ------------------------------------------------------------------
__global__ __launch_bounds__(256) void dwconv_v(const float* __restrict__ v,
                                                const float* __restrict__ w9,
                                                const float* __restrict__ cb,
                                                float* __restrict__ outv) {
  const int bc = blockIdx.z;
  const int b = bc >> 8, c = bc & 255;
  const int head = c >> 5, dd = c & 31;
  const size_t chan = (((size_t)(b * NHEAD + head)) * NN) * DHEAD + dd;
  const float* base = v + chan;
  __shared__ float t[18][20];
  const int tx0 = blockIdx.x * 16, ty0 = blockIdx.y * 16;
  for (int i = threadIdx.x; i < 18 * 18; i += 256) {
    const int iy = i / 18, ix = i - iy * 18;
    const int y = ty0 + iy - 1, x = tx0 + ix - 1;
    float val = 0.f;
    if (y >= 0 && y < HH && x >= 0 && x < WW2)
      val = base[(size_t)(y * WW2 + x) * DHEAD];
    t[iy][ix] = val;
  }
  __syncthreads();
  const int lx = threadIdx.x & 15, ly = threadIdx.x >> 4;
  const float* W = w9 + c * 9;
  float s = cb[c];
#pragma unroll
  for (int ky = 0; ky < 3; ++ky)
#pragma unroll
    for (int kx = 0; kx < 3; ++kx)
      s += t[ly + ky][lx + kx] * W[ky * 3 + kx];
  const int y = ty0 + ly, x = tx0 + lx;
  outv[chan + (size_t)(y * WW2 + x) * DHEAD] = s;
}

// ------------------------------------------------------------------
// att[b,n,h*32+dd] = scale * (q . kv[:,dd]) + q[dd]*convv[dd]  -> bf16
// grid: (N/256, h, B); kv tile staged in LDS per (b,h).
// ------------------------------------------------------------------
__global__ __launch_bounds__(256) void att_kernel(const float* __restrict__ q,
                                                  const float* __restrict__ kv,
                                                  const float* __restrict__ convv,
                                                  bf16_t* __restrict__ att) {
  const int b = blockIdx.z, h = blockIdx.y;
  const int bh = b * NHEAD + h;
  __shared__ float kvs[DHEAD * DHEAD];
  for (int i = threadIdx.x; i < DHEAD * DHEAD; i += 256)
    kvs[i] = kv[(size_t)bh * (DHEAD * DHEAD) + i];
  __syncthreads();
  const int n = blockIdx.x * 256 + threadIdx.x;
  const float* qrow = q     + ((size_t)bh * NN + n) * DHEAD;
  const float* cv   = convv + ((size_t)bh * NN + n) * DHEAD;
  float qr[32];
#pragma unroll
  for (int i = 0; i < 32; ++i) qr[i] = qrow[i];
  bf16_t* dst = att + ((size_t)(b * NN + n)) * CC + h * DHEAD;
  const float scale = 0.17677669529663687f;  // 32^-0.5
  for (int dd = 0; dd < 32; ++dd) {
    float s = 0.f;
#pragma unroll
    for (int kq = 0; kq < 32; ++kq) s += qr[kq] * kvs[kq * 32 + dd];
    dst[dd] = (bf16_t)(scale * s + qr[dd] * cv[dd]);
  }
}

// ------------------------------------------------------------------
// MLP depthwise 3x3 conv on hdn [B,N,hid] + exact GELU(hdn+dc) -> bf16
// grid: (8,8, B*hid)
// ------------------------------------------------------------------
__global__ __launch_bounds__(256) void mlp_dwconv_gelu(const float* __restrict__ hdn,
                                                       const float* __restrict__ w9,
                                                       const float* __restrict__ cb,
                                                       bf16_t* __restrict__ out) {
  const int bc = blockIdx.z;
  const int b = bc / HID, c = bc - b * HID;
  const float* base = hdn + (size_t)b * NN * HID + c;
  __shared__ float t[18][20];
  const int tx0 = blockIdx.x * 16, ty0 = blockIdx.y * 16;
  for (int i = threadIdx.x; i < 18 * 18; i += 256) {
    const int iy = i / 18, ix = i - iy * 18;
    const int y = ty0 + iy - 1, x = tx0 + ix - 1;
    float val = 0.f;
    if (y >= 0 && y < HH && x >= 0 && x < WW2)
      val = base[(size_t)(y * WW2 + x) * HID];
    t[iy][ix] = val;
  }
  __syncthreads();
  const int lx = threadIdx.x & 15, ly = threadIdx.x >> 4;
  const float* W = w9 + c * 9;
  float dc = cb[c];
#pragma unroll
  for (int ky = 0; ky < 3; ++ky)
#pragma unroll
    for (int kx = 0; kx < 3; ++kx)
      dc += t[ly + ky][lx + kx] * W[ky * 3 + kx];
  const float center = t[ly + 1][lx + 1];
  const float u = center + dc;
  const float gl = 0.5f * u * (1.0f + erff(u * 0.70710678118654752f));
  const int y = ty0 + ly, x = tx0 + lx;
  out[((size_t)b * NN + (size_t)(y * WW2 + x)) * HID + c] = (bf16_t)gl;
}

// ------------------------------------------------------------------
// Host orchestration
// ------------------------------------------------------------------
extern "C" void kernel_launch(void* const* d_in, const int* in_sizes, int n_in,
                              void* d_out, int out_size, void* d_ws, size_t ws_size,
                              hipStream_t stream) {
  (void)in_sizes; (void)n_in; (void)out_size; (void)ws_size;
  const float* x      = (const float*)d_in[0];
  const float* ln1_g  = (const float*)d_in[3];
  const float* ln1_b  = (const float*)d_in[4];
  const float* qkv_w  = (const float*)d_in[5];
  const float* qkv_b  = (const float*)d_in[6];
  const float* crpe_w = (const float*)d_in[7];
  const float* crpe_b = (const float*)d_in[8];
  const float* proj_w = (const float*)d_in[9];
  const float* proj_b = (const float*)d_in[10];
  const float* ln2_g  = (const float*)d_in[11];
  const float* ln2_b  = (const float*)d_in[12];
  const float* fc1_w  = (const float*)d_in[13];
  const float* fc1_b  = (const float*)d_in[14];
  const float* dconv_w= (const float*)d_in[15];
  const float* dconv_b= (const float*)d_in[16];
  const float* fc2_w  = (const float*)d_in[17];
  const float* fc2_b  = (const float*)d_in[18];
  float* out = (float*)d_out;

  // workspace carve-out (256B aligned); q/k/v and convv/att are contiguous
  // so hdn and hact can alias them after they are dead.
  char* ws = (char*)d_ws;
  size_t off = 0;
  auto take = [&](size_t bytes) {
    void* p = ws + off;
    off += (bytes + 255) & ~(size_t)255;
    return p;
  };
  bf16_t* wtQKV = (bf16_t*)take((size_t)CC * 3 * CC * sizeof(bf16_t));
  bf16_t* wtPRJ = (bf16_t*)take((size_t)CC * CC * sizeof(bf16_t));
  bf16_t* wtFC1 = (bf16_t*)take((size_t)CC * HID * sizeof(bf16_t));
  bf16_t* wtFC2 = (bf16_t*)take((size_t)HID * CC * sizeof(bf16_t));
  bf16_t* curA  = (bf16_t*)take((size_t)MM * CC * sizeof(bf16_t));
  float*  qbuf  = (float*)take((size_t)MM * CC * sizeof(float));   // [B,h,N,d]
  float*  kbuf  = (float*)take((size_t)MM * CC * sizeof(float));
  float*  vbuf  = (float*)take((size_t)MM * CC * sizeof(float));
  float*  kvbuf = (float*)take((size_t)BB * NHEAD * DHEAD * DHEAD * sizeof(float));
  float*  convv = (float*)take((size_t)MM * CC * sizeof(float));
  bf16_t* attb  = (bf16_t*)take((size_t)MM * CC * sizeof(bf16_t));
  float*  x1    = (float*)take((size_t)MM * CC * sizeof(float));
  float*  hdn   = qbuf;            // aliases q+k+v (201 MB), dead by then
  bf16_t* hact  = (bf16_t*)convv;  // aliases convv+att (100 MB), dead by then

  // 0) weight transposes (f32 -> bf16, WT[nc][k])
  transpose_w<<<(CC * 3 * CC + 255) / 256, 256, 0, stream>>>(qkv_w, wtQKV, CC, 3 * CC);
  transpose_w<<<(CC * CC + 255) / 256, 256, 0, stream>>>(proj_w, wtPRJ, CC, CC);
  transpose_w<<<(CC * HID + 255) / 256, 256, 0, stream>>>(fc1_w, wtFC1, CC, HID);
  transpose_w<<<(HID * CC + 255) / 256, 256, 0, stream>>>(fc2_w, wtFC2, HID, CC);

  // 1) LN1 -> bf16
  ln_kernel<<<MM / 8, 256, 0, stream>>>(x, ln1_g, ln1_b, curA);

  // 2) QKV GEMM (WMMA), split into q/k/v [B,h,N,d]
  gemm_bf16_wmma<2><<<dim3((3 * CC) / 64, MM / 128), 256, 0, stream>>>(
      curA, wtQKV, qkv_b, nullptr, nullptr, qbuf, kbuf, vbuf, CC, 3 * CC);

  // 3) softmax over tokens (in-place on k)
  softmax_tokens<<<BB * NHEAD * DHEAD, 256, 0, stream>>>(kbuf);

  // 4) kv = k_sm^T v (split-K + atomics)
  hipMemsetAsync(kvbuf, 0, (size_t)BB * NHEAD * DHEAD * DHEAD * sizeof(float), stream);
  kv_kernel<<<dim3(16, BB * NHEAD), 256, 0, stream>>>(kbuf, vbuf, kvbuf);

  // 5) CRPE depthwise conv of v
  dwconv_v<<<dim3(8, 8, BB * CC), 256, 0, stream>>>(vbuf, crpe_w, crpe_b, convv);

  // 6) att = scale*(q@kv) + q*conv_v  -> bf16 [B,N,C]
  att_kernel<<<dim3(NN / 256, NHEAD, BB), 256, 0, stream>>>(qbuf, kvbuf, convv, attb);

  // 7) proj GEMM (WMMA) + bias + residual(x) -> x1
  gemm_bf16_wmma<1><<<dim3(CC / 64, MM / 128), 256, 0, stream>>>(
      attb, wtPRJ, proj_b, x, x1, nullptr, nullptr, nullptr, CC, CC);

  // 8) LN2 -> bf16 (reuse curA)
  ln_kernel<<<MM / 8, 256, 0, stream>>>(x1, ln2_g, ln2_b, curA);

  // 9) fc1 GEMM (WMMA) + bias -> hdn f32 [B,N,hid]
  gemm_bf16_wmma<0><<<dim3(HID / 64, MM / 128), 256, 0, stream>>>(
      curA, wtFC1, fc1_b, nullptr, hdn, nullptr, nullptr, nullptr, CC, HID);

  // 10) depthwise conv + GELU(hdn + dc) -> bf16 hact
  mlp_dwconv_gelu<<<dim3(8, 8, BB * HID), 256, 0, stream>>>(hdn, dconv_w, dconv_b, hact);

  // 11) fc2 GEMM (WMMA) + bias + residual(x1) -> output
  gemm_bf16_wmma<1><<<dim3(CC / 64, MM / 128), 256, 0, stream>>>(
      hact, wtFC2, fc2_b, x1, out, nullptr, nullptr, nullptr, HID, CC);
}
